// PolymorphicNeuronLayer_712964571621
// MI455X (gfx1250) — matrix operand order repaired and checked by reading, not compiled, not run on hardware
//
#include <hip/hip_runtime.h>
#include <math.h>

typedef __attribute__((ext_vector_type(8)))  float  v8f;
typedef __attribute__((ext_vector_type(4)))  __bf16 v4bf;
typedef __attribute__((ext_vector_type(8)))  __bf16 v8bf;
typedef __attribute__((ext_vector_type(16))) __bf16 v16bf;

#define MAT_N 4096              // batch = out = in = 4096
#define TILE_M 128
#define TILE_N 128
#define TILE_K 32
#define LDS_K  40               // padded bf16 row stride: 16B-aligned, conflict-free
#define PLANE_ELEMS ((size_t)MAT_N * MAT_N)

struct BF16Pair { __bf16 hi; __bf16 lo; };

__device__ __forceinline__ BF16Pair split_bf16x3(float v)
{
    BF16Pair p;
    __bf16 h = (__bf16)v;              // RNE
    p.hi = h;
    p.lo = (__bf16)(v - (float)h);     // residual
    return p;
}

__device__ __forceinline__ v16bf load_frag2(const __bf16* c0, const __bf16* c1)
{
    v8bf a = *(const v8bf*)c0;
    v8bf b = *(const v8bf*)c1;
    return __builtin_shufflevector(a, b, 0,1,2,3,4,5,6,7,8,9,10,11,12,13,14,15);
}

// ---------------------------------------------------------------------------
// Pre-pass: split f32 -> bf16 hi/lo planes (bandwidth bound, runs once).
// blockIdx.y: 0 -> x planes, 1 -> W planes.
// ---------------------------------------------------------------------------
__global__ __launch_bounds__(256)
void split_planes_kernel(const float* __restrict__ x,
                         const float* __restrict__ W,
                         __bf16* __restrict__ ws)
{
    const size_t t = (size_t)blockIdx.x * 256 + threadIdx.x;   // float4 index
    const int mat = blockIdx.y;                                // 0=x, 1=W
    const float* src = mat ? W : x;
    __bf16* hiP = ws + (size_t)(2 * mat) * PLANE_ELEMS;
    __bf16* loP = ws + (size_t)(2 * mat + 1) * PLANE_ELEMS;

    float4 v = *(const float4*)(src + t * 4);
    v4bf hi, lo;
    BF16Pair p0 = split_bf16x3(v.x), p1 = split_bf16x3(v.y);
    BF16Pair p2 = split_bf16x3(v.z), p3 = split_bf16x3(v.w);
    hi[0] = p0.hi; lo[0] = p0.lo;  hi[1] = p1.hi; lo[1] = p1.lo;
    hi[2] = p2.hi; lo[2] = p2.lo;  hi[3] = p3.hi; lo[3] = p3.lo;
    *(v4bf*)(hiP + t * 4) = hi;
    *(v4bf*)(loP + t * 4) = lo;
}

// ---------------------------------------------------------------------------
// Shared compute core: 3-term bf16x3 accumulation from LDS tiles.
// ---------------------------------------------------------------------------
__device__ __forceinline__ void wmma_slab(const __bf16* AsHi, const __bf16* AsLo,
                                          const __bf16* BsHi, const __bf16* BsLo,
                                          int wm, int wn, int lhalf, int lmod,
                                          v8f acc[4][2])
{
    v16bf bH[2], bL[2];
#pragma unroll
    for (int j = 0; j < 2; ++j) {
        const int nrow = wn * 32 + j * 16 + lmod;
        const __bf16* ph = BsHi + nrow * LDS_K + 16 * lhalf;
        const __bf16* pl = BsLo + nrow * LDS_K + 16 * lhalf;
        bH[j] = load_frag2(ph, ph + 8);
        bL[j] = load_frag2(pl, pl + 8);
    }
#pragma unroll
    for (int i = 0; i < 4; ++i) {
        const int mrow = wm * 64 + i * 16 + lmod;
        const __bf16* ph = AsHi + mrow * LDS_K;
        const __bf16* pl = AsLo + mrow * LDS_K;
        v16bf aH = load_frag2(ph + 16 * lhalf, ph + 16 * lhalf + 8);
        v16bf aL = load_frag2(pl + 16 * lhalf, pl + 16 * lhalf + 8);
#pragma unroll
        for (int j = 0; j < 2; ++j) {
            acc[i][j] = __builtin_amdgcn_wmma_f32_16x16x32_bf16(
                false, aL, false, bH[j], (short)0, acc[i][j], false, false);
            acc[i][j] = __builtin_amdgcn_wmma_f32_16x16x32_bf16(
                false, aH, false, bL[j], (short)0, acc[i][j], false, false);
            acc[i][j] = __builtin_amdgcn_wmma_f32_16x16x32_bf16(
                false, aH, false, bH[j], (short)0, acc[i][j], false, false);
        }
    }
}

__device__ __forceinline__ void store_epilogue(float* __restrict__ y,
                                               const float* __restrict__ bias,
                                               int blockM, int blockN,
                                               int wm, int wn, int lhalf, int lmod,
                                               v8f acc[4][2])
{
#pragma unroll
    for (int j = 0; j < 2; ++j) {
        const int col = blockN + wn * 32 + j * 16 + lmod;
        const float bv = bias[col];
#pragma unroll
        for (int i = 0; i < 4; ++i) {
            const int rowBase = blockM + wm * 64 + i * 16 + 8 * lhalf;
#pragma unroll
            for (int v = 0; v < 8; ++v)
                y[(size_t)(rowBase + v) * MAT_N + col] = acc[i][j][v] + bv;
        }
    }
}

// ---------------------------------------------------------------------------
// Main GEMM from pre-split bf16 planes: zero conversion VALU in the hot loop.
// ---------------------------------------------------------------------------
__global__ __launch_bounds__(256)
void gemm_bias_wmma_planes_kernel(const __bf16* __restrict__ ws,
                                  const float* __restrict__ bias,
                                  float* __restrict__ y)
{
    __shared__ __bf16 AsHi[TILE_M * LDS_K];
    __shared__ __bf16 AsLo[TILE_M * LDS_K];
    __shared__ __bf16 BsHi[TILE_N * LDS_K];
    __shared__ __bf16 BsLo[TILE_N * LDS_K];

    const __bf16* xHi = ws;
    const __bf16* xLo = ws + PLANE_ELEMS;
    const __bf16* wHi = ws + 2 * PLANE_ELEMS;
    const __bf16* wLo = ws + 3 * PLANE_ELEMS;

    const int tid   = threadIdx.x;
    const int wave  = tid >> 5;
    const int lane  = tid & 31;
    const int wm    = wave >> 2;
    const int wn    = wave & 3;
    const int lhalf = lane >> 4;
    const int lmod  = lane & 15;

    const int blockM = blockIdx.y * TILE_M;
    const int blockN = blockIdx.x * TILE_N;

    v8f acc[4][2];
#pragma unroll
    for (int i = 0; i < 4; ++i)
#pragma unroll
        for (int j = 0; j < 2; ++j)
            acc[i][j] = (v8f)0.0f;

    for (int k0 = 0; k0 < MAT_N; k0 += TILE_K) {
        // stage 4 bf16 slabs (128x32 each) as raw 16B copies
#pragma unroll
        for (int i = 0; i < 2; ++i) {
            int id  = tid + i * 256;          // 0..511
            int row = id >> 2;                // 0..127
            int cv  = (id & 3) << 3;          // 0,8,16,24
            const size_t gA = (size_t)(blockM + row) * MAT_N + k0 + cv;
            const size_t gB = (size_t)(blockN + row) * MAT_N + k0 + cv;
            const int   lo  = row * LDS_K + cv;
            *(v8bf*)(AsHi + lo) = *(const v8bf*)(xHi + gA);
            *(v8bf*)(AsLo + lo) = *(const v8bf*)(xLo + gA);
            *(v8bf*)(BsHi + lo) = *(const v8bf*)(wHi + gB);
            *(v8bf*)(BsLo + lo) = *(const v8bf*)(wLo + gB);
        }
        __syncthreads();

        if (k0 + TILE_K < MAT_N) {
            __builtin_prefetch(xHi + (size_t)(blockM + (tid >> 2)) * MAT_N + k0 + TILE_K, 0, 3);
            __builtin_prefetch(wHi + (size_t)(blockN + (tid >> 2)) * MAT_N + k0 + TILE_K, 0, 3);
        }

        wmma_slab(AsHi, AsLo, BsHi, BsLo, wm, wn, lhalf, lmod, acc);
        __syncthreads();
    }

    store_epilogue(y, bias, blockM, blockN, wm, wn, lhalf, lmod, acc);
}

// ---------------------------------------------------------------------------
// Fallback GEMM (inline split) for when d_ws can't hold the bf16 planes.
// ---------------------------------------------------------------------------
__global__ __launch_bounds__(256)
void gemm_bias_wmma_inline_kernel(const float* __restrict__ x,
                                  const float* __restrict__ W,
                                  const float* __restrict__ bias,
                                  float* __restrict__ y)
{
    __shared__ __bf16 AsHi[TILE_M * LDS_K];
    __shared__ __bf16 AsLo[TILE_M * LDS_K];
    __shared__ __bf16 BsHi[TILE_N * LDS_K];
    __shared__ __bf16 BsLo[TILE_N * LDS_K];

    const int tid   = threadIdx.x;
    const int wave  = tid >> 5;
    const int lane  = tid & 31;
    const int wm    = wave >> 2;
    const int wn    = wave & 3;
    const int lhalf = lane >> 4;
    const int lmod  = lane & 15;

    const int blockM = blockIdx.y * TILE_M;
    const int blockN = blockIdx.x * TILE_N;

    v8f acc[4][2];
#pragma unroll
    for (int i = 0; i < 4; ++i)
#pragma unroll
        for (int j = 0; j < 2; ++j)
            acc[i][j] = (v8f)0.0f;

    for (int k0 = 0; k0 < MAT_N; k0 += TILE_K) {
#pragma unroll
        for (int i = 0; i < 4; ++i) {
            int id  = tid + i * 256;
            int row = id >> 3;
            int cv  = (id & 7) << 2;
            float4 va = *(const float4*)(x + (size_t)(blockM + row) * MAT_N + k0 + cv);
            float4 vb = *(const float4*)(W + (size_t)(blockN + row) * MAT_N + k0 + cv);
            v4bf ahi, alo, bhi, blo;
            {
                BF16Pair p0 = split_bf16x3(va.x), p1 = split_bf16x3(va.y);
                BF16Pair p2 = split_bf16x3(va.z), p3 = split_bf16x3(va.w);
                ahi[0] = p0.hi; alo[0] = p0.lo;  ahi[1] = p1.hi; alo[1] = p1.lo;
                ahi[2] = p2.hi; alo[2] = p2.lo;  ahi[3] = p3.hi; alo[3] = p3.lo;
            }
            {
                BF16Pair p0 = split_bf16x3(vb.x), p1 = split_bf16x3(vb.y);
                BF16Pair p2 = split_bf16x3(vb.z), p3 = split_bf16x3(vb.w);
                bhi[0] = p0.hi; blo[0] = p0.lo;  bhi[1] = p1.hi; blo[1] = p1.lo;
                bhi[2] = p2.hi; blo[2] = p2.lo;  bhi[3] = p3.hi; blo[3] = p3.lo;
            }
            *(v4bf*)(AsHi + row * LDS_K + cv) = ahi;
            *(v4bf*)(AsLo + row * LDS_K + cv) = alo;
            *(v4bf*)(BsHi + row * LDS_K + cv) = bhi;
            *(v4bf*)(BsLo + row * LDS_K + cv) = blo;
        }
        __syncthreads();
        wmma_slab(AsHi, AsLo, BsHi, BsLo, wm, wn, lhalf, lmod, acc);
        __syncthreads();
    }

    store_epilogue(y, bias, blockM, blockN, wm, wn, lhalf, lmod, acc);
}

// One thread per output column: column mean -> activation index -> rewrite column.
__global__ __launch_bounds__(256)
void colmean_activation_kernel(float* __restrict__ y)
{
    const int o = blockIdx.x * 256 + threadIdx.x;
    float s = 0.0f;
    for (int r = 0; r < MAT_N; ++r)
        s += y[(size_t)r * MAT_N + o];
    const float mean = s * (1.0f / (float)MAT_N);
    float m = mean - 3.0f * floorf(mean * (1.0f / 3.0f));
    int idx = (int)m;
    if (idx > 2) idx = 2;
    if (idx < 0) idx = 0;

    for (int r = 0; r < MAT_N; ++r) {
        const float v  = y[(size_t)r * MAT_N + o];
        const float re = fmaxf(v, 0.0f);
        const float th = tanhf(v);
        const float sg = 1.0f / (1.0f + expf(-v));
        const float outv = (idx == 0) ? re : ((idx == 1) ? th : sg);
        y[(size_t)r * MAT_N + o] = outv;
    }
}

extern "C" void kernel_launch(void* const* d_in, const int* in_sizes, int n_in,
                              void* d_out, int out_size, void* d_ws, size_t ws_size,
                              hipStream_t stream)
{
    const float* x = (const float*)d_in[0];   // (4096, 4096)
    const float* W = (const float*)d_in[1];   // (4096, 4096)
    const float* b = (const float*)d_in[2];   // (4096,)
    float* y = (float*)d_out;                 // (4096, 4096)

    const size_t need = 4 * PLANE_ELEMS * sizeof(__bf16);   // 128 MB of bf16 planes
    dim3 grid(MAT_N / TILE_N, MAT_N / TILE_M);

    if (ws_size >= need) {
        __bf16* ws = (__bf16*)d_ws;
        dim3 sgrid((unsigned)(PLANE_ELEMS / 4 / 256), 2);
        split_planes_kernel<<<sgrid, 256, 0, stream>>>(x, W, ws);
        gemm_bias_wmma_planes_kernel<<<grid, 256, 0, stream>>>(ws, b, y);
    } else {
        gemm_bias_wmma_inline_kernel<<<grid, 256, 0, stream>>>(x, W, b, y);
    }
    colmean_activation_kernel<<<MAT_N / 256, 256, 0, stream>>>(y);
}
